// Model_64536178589720
// MI455X (gfx1250) — compile-verified, matrix operand order checked by reading, not run on hardware
//
#include <hip/hip_runtime.h>
#include <math.h>

// ---------------- CDNA5 WMMA types ----------------
typedef __attribute__((ext_vector_type(16))) __bf16 v16bf;
typedef __attribute__((ext_vector_type(8)))  float  v8f;

struct alignas(16) U4 { unsigned int x, y, z, w; };
union Frag { v16bf v; U4 q[2]; };

#define GRID_H   0.2f
#define LAMBD_F  0.001f
#define RSQRT96  0.1020620726159658f

__device__ __forceinline__ unsigned short f2bf(float f) {
  unsigned int u = __float_as_uint(f);
  u += 0x7FFFu + ((u >> 16) & 1u);   // round-to-nearest-even
  return (unsigned short)(u >> 16);
}

// silu + 13 cubic B-spline bases (Cox-de Boor on uniform grid h=0.2 over [-1.6,1.6])
__device__ __forceinline__ void kan_expand(float x, float* e) {
  e[0] = x / (1.0f + __expf(-x));
  float b[16];
#pragma unroll
  for (int j = 0; j < 16; ++j) {
    float g0 = (float)(j - 3) * GRID_H - 1.0f;
    float g1 = (float)(j - 2) * GRID_H - 1.0f;
    b[j] = (x >= g0 && x < g1) ? 1.0f : 0.0f;
  }
#pragma unroll
  for (int k = 1; k <= 3; ++k) {
    float inv = 1.0f / (GRID_H * (float)k);
#pragma unroll
    for (int j = 0; j < 16 - k; ++j) {
      float gj   = (float)(j - 3) * GRID_H - 1.0f;      // g[j]
      float gjk1 = (float)(j + k - 2) * GRID_H - 1.0f;  // g[j+k+1]
      b[j] = (x - gj) * inv * b[j] + (gjk1 - x) * inv * b[j + 1];
    }
  }
#pragma unroll
  for (int c = 0; c < 13; ++c) e[1 + c] = b[c];
}

// ---------------- weight packing: [o][i*14 + {bw, sw*ss}] as bf16 ----------------
__global__ void pack_kernel(const float* __restrict__ bw, const float* __restrict__ sw,
                            const float* __restrict__ ss, unsigned short* __restrict__ wp,
                            int total) {
  int idx = blockIdx.x * blockDim.x + threadIdx.x;
  if (idx >= total) return;
  float scale = ss[idx];
  unsigned short* o = wp + (size_t)idx * 14;
  o[0] = f2bf(bw[idx]);
  const float* s = sw + (size_t)idx * 13;
#pragma unroll
  for (int c = 0; c < 13; ++c) o[1 + c] = f2bf(s[c] * scale);
}

// ---------------- ortho DFT twiddles (exact mod-96 reduction) ----------------
__global__ void twiddle_kernel(float* __restrict__ tc, float* __restrict__ ts) {
  int idx = blockIdx.x * blockDim.x + threadIdx.x;
  if (idx >= 49 * 96) return;
  int k = idx / 96, t = idx % 96;
  int m = (k * t) % 96;
  float ang = 6.283185307179586f * (float)m / 96.0f;
  float s, c;
  sincosf(ang, &s, &c);
  tc[idx] = c * RSQRT96;
  ts[idx] = s * RSQRT96;
}

// xe[b][n][t][d] = x[b][t][n] * emb[d]
__global__ void embed_kernel(const float* __restrict__ x, const float* __restrict__ emb,
                             float* __restrict__ xe) {
  int idx = blockIdx.x * blockDim.x + threadIdx.x;   // < 16*32*96*128
  int d = idx & 127;
  int tn = idx >> 7;
  int t = tn % 96;
  int bn = tn / 96;
  int n = bn & 31, b = bn >> 5;
  xe[idx] = x[((b * 96) + t) * 32 + n] * emb[d];
}

// rfft (ortho) over T for each (bn,d): one thread per (bn,k,d)
__global__ void rfft_kernel(const float* __restrict__ xe, const float* __restrict__ tc,
                            const float* __restrict__ ts, float* __restrict__ xr,
                            float* __restrict__ xi) {
  int idx = blockIdx.x * blockDim.x + threadIdx.x;   // < 512*49*128
  int d = idx & 127;
  int k = (idx >> 7) % 49;
  int bn = idx / (49 * 128);
  const float* src = xe + (size_t)bn * 96 * 128 + d;
  const float* c = tc + k * 96;
  const float* s = ts + k * 96;
  float ar = 0.0f, ai = 0.0f;
  for (int t = 0; t < 96; ++t) {
    float v = src[(size_t)t * 128];
    ar = fmaf(v, c[t], ar);
    ai = fmaf(-v, s[t], ai);
  }
  xr[idx] = ar;
  xi[idx] = ai;
}

// irfft (ortho, c2r: imag of k=0,48 ignored) fused with h = x_freq + x_time + 2*xe (in place)
__global__ void irfft_compose_kernel(const float* __restrict__ yr, const float* __restrict__ yi,
                                     const float* __restrict__ xt, const float* __restrict__ tc,
                                     const float* __restrict__ ts, float* __restrict__ xe_h) {
  int idx = blockIdx.x * blockDim.x + threadIdx.x;   // < 512*96*128
  int d = idx & 127;
  int t = (idx >> 7) % 96;
  int bn = idx / (96 * 128);
  const float* r  = yr + (size_t)bn * 49 * 128 + d;
  const float* im = yi + (size_t)bn * 49 * 128 + d;
  float acc = 0.0f;
  for (int k = 0; k < 49; ++k) {
    float w = (k == 0 || k == 48) ? 1.0f : 2.0f;
    float cc = tc[k * 96 + t], sc = ts[k * 96 + t];
    acc += w * (r[(size_t)k * 128] * cc - im[(size_t)k * 128] * sc);
  }
  xe_h[idx] = acc + xt[idx] + 2.0f * xe_h[idx];
}

// ---------------- fused KANLinear GEMM with WMMA bf16 ----------------
// out[R,OUT] = expand14(in[R,IN]) @ wpack[OUT, IN*14]^T ; mode 1 => softshrink
// Block: 128 thr (4 waves). WG tile: 64 rows x 64 cols; each wave: 64x16 (4 accs).
__global__ __launch_bounds__(128) void kan_gemm_kernel(
    const float* __restrict__ in, const unsigned short* __restrict__ wpack,
    float* __restrict__ out, int R, int IN, int OUT, int mode) {
  __shared__ unsigned short a_lds[64 * 224];   // 64 rows x (16 inputs * 14) bf16
  const int tid  = threadIdx.x;
  const int lane = tid & 31;
  const int wv   = tid >> 5;
  const int half = lane >> 4;
  const int ln   = lane & 15;
  const int row0 = blockIdx.x * 64;
  const int col0 = blockIdx.y * 64 + wv * 16;
  const bool active = (col0 < OUT);

  v8f acc[4];
#pragma unroll
  for (int mt = 0; mt < 4; ++mt)
#pragma unroll
    for (int r = 0; r < 8; ++r) acc[mt][r] = 0.0f;

  const size_t wstride = (size_t)IN * 14;
  const unsigned short* wrow = wpack + (size_t)(col0 + ln) * wstride;
  const int ngroups = IN >> 4;

  for (int g = 0; g < ngroups; ++g) {
    __syncthreads();
    // cooperative expansion: 64 rows x 16 inputs -> 14 bf16 each
#pragma unroll
    for (int rep = 0; rep < 8; ++rep) {
      int p  = tid + rep * 128;   // 0..1023
      int rr = p >> 4;
      int ii = p & 15;
      float xval = in[(size_t)(row0 + rr) * IN + (g * 16 + ii)];
      float e[14];
      kan_expand(xval, e);
      unsigned int* dst = (unsigned int*)a_lds + (rr * 112 + ii * 7);
#pragma unroll
      for (int c = 0; c < 7; ++c) {
        unsigned int lo = f2bf(e[2 * c]);
        unsigned int hi = f2bf(e[2 * c + 1]);
        dst[c] = lo | (hi << 16);
      }
    }
    __syncthreads();

    if (active) {
      const int kg = g * 224;
#pragma unroll
      for (int s = 0; s < 7; ++s) {
        const int kb = kg + s * 32;
        Frag bf;
        bf.q[0] = *(const U4*)(wrow + kb + 8 * half);
        bf.q[1] = *(const U4*)(wrow + kb + 16 + 8 * half);
#pragma unroll
        for (int mt = 0; mt < 4; ++mt) {
          Frag af;
          const unsigned short* ap = a_lds + ((mt * 16 + ln) * 224 + s * 32 + 8 * half);
          af.q[0] = *(const U4*)(ap);
          af.q[1] = *(const U4*)(ap + 16);
          acc[mt] = __builtin_amdgcn_wmma_f32_16x16x32_bf16(
              false, af.v, false, bf.v, (short)0, acc[mt], false, false);
        }
      }
      if (g + 1 < ngroups)
        __builtin_prefetch(wrow + kg + 224, 0, 0);   // global_prefetch_b8
    }
  }

  if (active) {
#pragma unroll
    for (int mt = 0; mt < 4; ++mt) {
#pragma unroll
      for (int r = 0; r < 8; ++r) {
        float v = acc[mt][r];
        if (mode == 1)
          v = (v > LAMBD_F) ? v - LAMBD_F : ((v < -LAMBD_F) ? v + LAMBD_F : 0.0f);
        int row = row0 + mt * 16 + r + 8 * half;
        out[(size_t)row * OUT + (col0 + ln)] = v;
      }
    }
  }
}

// out[b][p][n] = fcout[b*32+n][p]
__global__ void out_transpose_kernel(const float* __restrict__ fco, float* __restrict__ out) {
  int idx = blockIdx.x * blockDim.x + threadIdx.x;
  if (idx >= 16 * 96 * 32) return;
  int n = idx & 31;
  int p = (idx >> 5) % 96;
  int b = idx / (96 * 32);
  out[idx] = fco[(size_t)(b * 32 + n) * 96 + p];
}

extern "C" void kernel_launch(void* const* d_in, const int* in_sizes, int n_in,
                              void* d_out, int out_size, void* d_ws, size_t ws_size,
                              hipStream_t stream) {
  (void)in_sizes; (void)n_in; (void)out_size; (void)ws_size;
  const float* x     = (const float*)d_in[0];
  const float* emb   = (const float*)d_in[1];
  const float* k0_bw = (const float*)d_in[2];
  const float* k0_sw = (const float*)d_in[3];
  const float* k0_ss = (const float*)d_in[4];
  const float* k1_bw = (const float*)d_in[5];
  const float* k1_sw = (const float*)d_in[6];
  const float* k1_ss = (const float*)d_in[7];
  const float* t0_bw = (const float*)d_in[8];
  const float* t0_sw = (const float*)d_in[9];
  const float* t0_ss = (const float*)d_in[10];
  const float* t1_bw = (const float*)d_in[11];
  const float* t1_sw = (const float*)d_in[12];
  const float* t1_ss = (const float*)d_in[13];
  const float* f0_bw = (const float*)d_in[14];
  const float* f0_sw = (const float*)d_in[15];
  const float* f0_ss = (const float*)d_in[16];
  const float* f1_bw = (const float*)d_in[17];
  const float* f1_sw = (const float*)d_in[18];
  const float* f1_ss = (const float*)d_in[19];
  float* out = (float*)d_out;

  // ---- workspace carve ----
  char* base = (char*)d_ws;
  size_t off = 0;
  auto carve = [&](size_t bytes) -> void* {
    void* p = base + off;
    off += (bytes + 255) & ~(size_t)255;
    return p;
  };
  float* xe  = (float*)carve(6291456ull * 4);   // [512][96][128], later holds h
  float* xt  = (float*)carve(6291456ull * 4);   // x_time
  float* xfr = (float*)carve(3211264ull * 4);   // [512][49][128], later yr
  float* xfi = (float*)carve(3211264ull * 4);   // later yi
  float* hid = (float*)carve(49152ull * 256 * 4);
  float* fco = (float*)carve(512ull * 96 * 4);
  float* tc  = (float*)carve(4704ull * 4);
  float* ts  = (float*)carve(4704ull * 4);
  unsigned short* wk0 = (unsigned short*)carve(458752ull * 2);    // 256x128x14
  unsigned short* wk1 = (unsigned short*)carve(458752ull * 2);    // 128x256x14
  unsigned short* wt0 = (unsigned short*)carve(458752ull * 2);
  unsigned short* wt1 = (unsigned short*)carve(458752ull * 2);
  unsigned short* wf0 = (unsigned short*)carve(44040192ull * 2);  // 256x12288x14
  unsigned short* wf1 = (unsigned short*)carve(344064ull * 2);    // 96x256x14

  // ---- pack weights (bf16 combined base+spline) ----
  pack_kernel<<<(32768 + 255) / 256, 256, 0, stream>>>(k0_bw, k0_sw, k0_ss, wk0, 32768);
  pack_kernel<<<(32768 + 255) / 256, 256, 0, stream>>>(k1_bw, k1_sw, k1_ss, wk1, 32768);
  pack_kernel<<<(32768 + 255) / 256, 256, 0, stream>>>(t0_bw, t0_sw, t0_ss, wt0, 32768);
  pack_kernel<<<(32768 + 255) / 256, 256, 0, stream>>>(t1_bw, t1_sw, t1_ss, wt1, 32768);
  pack_kernel<<<(3145728 + 255) / 256, 256, 0, stream>>>(f0_bw, f0_sw, f0_ss, wf0, 3145728);
  pack_kernel<<<(24576 + 255) / 256, 256, 0, stream>>>(f1_bw, f1_sw, f1_ss, wf1, 24576);

  twiddle_kernel<<<(4704 + 127) / 128, 128, 0, stream>>>(tc, ts);
  embed_kernel<<<24576, 256, 0, stream>>>(x, emb, xe);
  rfft_kernel<<<12544, 256, 0, stream>>>(xe, tc, ts, xfr, xfi);

  // frequency-domain KANs (25088 rows), softshrink on second layer
  kan_gemm_kernel<<<dim3(392, 4), 128, 0, stream>>>(xfr, wk0, hid, 25088, 128, 256, 0);
  kan_gemm_kernel<<<dim3(392, 2), 128, 0, stream>>>(hid, wk1, xfr, 25088, 256, 128, 1);
  kan_gemm_kernel<<<dim3(392, 4), 128, 0, stream>>>(xfi, wk0, hid, 25088, 128, 256, 0);
  kan_gemm_kernel<<<dim3(392, 2), 128, 0, stream>>>(hid, wk1, xfi, 25088, 256, 128, 1);

  // time-domain KAN (49152 rows)
  kan_gemm_kernel<<<dim3(768, 4), 128, 0, stream>>>(xe, wt0, hid, 49152, 128, 256, 0);
  kan_gemm_kernel<<<dim3(768, 2), 128, 0, stream>>>(hid, wt1, xt, 49152, 256, 128, 0);

  // irfft + h = x_freq + x_time + 2*xe (in place over xe)
  irfft_compose_kernel<<<24576, 256, 0, stream>>>(xfr, xfi, xt, tc, ts, xe);

  // fc KAN over flattened [512, 12288]
  kan_gemm_kernel<<<dim3(8, 4), 128, 0, stream>>>(xe, wf0, hid, 512, 12288, 256, 0);
  kan_gemm_kernel<<<dim3(8, 2), 128, 0, stream>>>(hid, wf1, fco, 512, 256, 96, 0);

  out_transpose_kernel<<<192, 256, 0, stream>>>(fco, out);
}
